// LearnableCost_1460288881502
// MI455X (gfx1250) — matrix operand order repaired and checked by reading, not compiled, not run on hardware
//
#include <hip/hip_runtime.h>

// LearnableCost for MI455X (gfx1250, wave32, WMMA).
// Dominant cost: h += |x_n - y_m| @ W1d  (B*N*M*D*H*2 = 34.4 GFLOP) -> f16 WMMA, f32 acc.
// Workspace layout (needs ~2.13 MB):
//   [0, 1MB)      preX[b,n,h] = x . W1x            (f32)
//   [1MB, 2MB)    preY[b,m,h] = y . W1y + b1       (f32)
//   [2MB, +32KB)  w1d16[h,k]  = W1d^T as _Float16  (B-operand source)

typedef __attribute__((ext_vector_type(16))) _Float16 v16h;
typedef __attribute__((ext_vector_type(8)))  _Float16 v8h;
typedef __attribute__((ext_vector_type(8)))  float    v8f;

#define Bn 4
#define Nn 512
#define Mn 512
#define Dn 128
#define Hn 128
#define KP 136   // padded row stride (halves) for W1d in LDS: 272B == 4 banks mod 64
#define XP 132   // padded row stride (floats) for x/y/pre tiles: 528B == 4 banks mod 64

// |a-b| without bit-pattern abs (avoids gfx1250 ISel crash on vectorized
// bitcast/and fabs idiom): lowers to v_sub_f32 + v_max_f32 (neg as src-mod).
__device__ __forceinline__ float absdiff(float a, float b) {
  return fmaxf(a - b, b - a);
}

// ---------------- Phase 1a: preX / preY small GEMMs ----------------
// One block per row (128 threads = H). rows: [R][128] f32, W: [128][128] f32 (pre-offset).
__global__ void pre_gemm_kernel(const float* __restrict__ rows,
                                const float* __restrict__ W,
                                const float* __restrict__ bias,
                                float* __restrict__ outp, int useBias) {
  __shared__ float s[128];
  const int row = blockIdx.x;
  const int h = threadIdx.x;
  s[h] = rows[(size_t)row * 128 + h];
  __syncthreads();
  float acc = useBias ? bias[h] : 0.f;
#pragma unroll 8
  for (int d = 0; d < 128; ++d) acc = fmaf(s[d], W[d * 128 + h], acc);
  outp[(size_t)row * 128 + h] = acc;
}

// ---------------- Phase 1b: W1d -> f16, transposed to [h][k] ----------------
__global__ void w1d_cvt_kernel(const float* __restrict__ W1, _Float16* __restrict__ w1d16) {
  int idx = blockIdx.x * 256 + threadIdx.x;   // 16384 total
  int h = idx >> 7, k = idx & 127;
  w1d16[idx] = (_Float16)W1[(2 * Dn + k) * Hn + h];
}

// ---------------- Phase 2: main fused kernel ----------------
// Block: 256 threads (8 waves). Tile: 16 n  x  8 m (one m per wave).
// Per wave: 32x v_wmma_f32_16x16x32_f16 building h[16 pairs][128], then fused
// LayerNorm + ReLU + W2 head + ReLU epilogue with cross-lane butterflies.
__global__ __launch_bounds__(256) void cost_kernel(
    const float* __restrict__ x, const float* __restrict__ y,
    const float* __restrict__ gamma, const float* __restrict__ beta,
    const float* __restrict__ W2, const float* __restrict__ b2,
    const float* __restrict__ preX, const float* __restrict__ preY,
    const _Float16* __restrict__ w1d16, float* __restrict__ out) {
  __shared__ _Float16 sW[Hn * KP];   // 34816 B
  __shared__ float sX[16 * XP];      // 8448 B
  __shared__ float sPX[16 * XP];     // 8448 B
  __shared__ float sY[8 * XP];       // 4224 B
  __shared__ float sPY[8 * XP];      // 4224 B  (total ~60 KB)

  const int t    = threadIdx.x;
  const int wv   = t >> 5;          // wave id -> local m
  const int lane = t & 31;
  const int r    = lane & 15;       // column / row-in-half index
  const int hh   = lane >> 4;       // lane half (K/row-half select)

  const int bz = blockIdx.z;
  const int n0 = blockIdx.y * 16;
  const int m0 = blockIdx.x * 8;
  const int m  = m0 + wv;

  // ---- stage LDS (whole block cooperates) ----
  {
    const uint4* src = (const uint4*)w1d16;   // packed [h*128+k], 16B chunks
#pragma unroll
    for (int i = 0; i < 8; ++i) {
      int c = t + i * 256;                    // 2048 chunks
      int h = c >> 4, kc = c & 15;
      *(uint4*)(&sW[h * KP + kc * 8]) = src[c];
    }
    const float4* xs = (const float4*)(x    + (size_t)(bz * Nn + n0) * Dn);
    const float4* ps = (const float4*)(preX + (size_t)(bz * Nn + n0) * Hn);
#pragma unroll
    for (int i = 0; i < 2; ++i) {
      int c = t + i * 256;                    // 512 float4
      int rr = c >> 5, kc = c & 31;
      *(float4*)(&sX[rr * XP + kc * 4])  = xs[c];
      *(float4*)(&sPX[rr * XP + kc * 4]) = ps[c];
    }
    const float4* ys = (const float4*)(y    + (size_t)(bz * Mn + m0) * Dn);
    const float4* qs = (const float4*)(preY + (size_t)(bz * Mn + m0) * Hn);
    {
      int rr = t >> 5, kc = t & 31;           // 256 float4 each
      *(float4*)(&sY[rr * XP + kc * 4])  = ys[t];
      *(float4*)(&sPY[rr * XP + kc * 4]) = qs[t];
    }
  }
  __syncthreads();

  // epilogue coefficients for this lane: h = c*16 + r
  float gc[8], bc[8], wc[8];
#pragma unroll
  for (int c = 0; c < 8; ++c) {
    int h = c * 16 + r;
    gc[c] = gamma[h]; bc[c] = beta[h]; wc[c] = W2[h];
  }
  const float b2v = b2[0];

  v8f acc[8] = {};   // 8 column tiles of H (c*16..c*16+15), f32 C/D layout

  // ---- main WMMA loop: D=128 as 4 chunks of K=32 ----
#pragma unroll
  for (int kk = 0; kk < 4; ++kk) {
    // A-operand (16-bit A 16x32 layout): lane(r,hh) holds row r,
    // K = kk*32 + hh*8 + {0..7} in elems 0..7 and +16 in elems 8..15.
    const int kb = kk * 32 + hh * 8;
    const float* xr = &sX[r * XP + kb];
    const float* yr = &sY[wv * XP + kb];
    float4 xa = *(const float4*)(xr);
    float4 xb = *(const float4*)(xr + 4);
    float4 xc = *(const float4*)(xr + 16);
    float4 xd = *(const float4*)(xr + 20);
    float4 ya = *(const float4*)(yr);
    float4 yb = *(const float4*)(yr + 4);
    float4 yc = *(const float4*)(yr + 16);
    float4 yd = *(const float4*)(yr + 20);
    v16h A;
    A[0]  = (_Float16)absdiff(xa.x, ya.x); A[1]  = (_Float16)absdiff(xa.y, ya.y);
    A[2]  = (_Float16)absdiff(xa.z, ya.z); A[3]  = (_Float16)absdiff(xa.w, ya.w);
    A[4]  = (_Float16)absdiff(xb.x, yb.x); A[5]  = (_Float16)absdiff(xb.y, yb.y);
    A[6]  = (_Float16)absdiff(xb.z, yb.z); A[7]  = (_Float16)absdiff(xb.w, yb.w);
    A[8]  = (_Float16)absdiff(xc.x, yc.x); A[9]  = (_Float16)absdiff(xc.y, yc.y);
    A[10] = (_Float16)absdiff(xc.z, yc.z); A[11] = (_Float16)absdiff(xc.w, yc.w);
    A[12] = (_Float16)absdiff(xd.x, yd.x); A[13] = (_Float16)absdiff(xd.y, yd.y);
    A[14] = (_Float16)absdiff(xd.z, yd.z); A[15] = (_Float16)absdiff(xd.w, yd.w);

#pragma unroll
    for (int c = 0; c < 8; ++c) {
      // B-operand (32x16): lane holds column h = c*16 + r,
      // K = kk*32 + hh*8 + {0..7} in elems 0..7 and +16 in elems 8..15.
      const _Float16* bp = &sW[(c * 16 + r) * KP + kk * 32 + hh * 8];
      v8h blo = *(const v8h*)bp;
      v8h bhi = *(const v8h*)(bp + 16);
      v16h Bf = __builtin_shufflevector(blo, bhi,
                 0,1,2,3,4,5,6,7,8,9,10,11,12,13,14,15);
      acc[c] = __builtin_amdgcn_wmma_f32_16x16x32_f16(
          false, A, false, Bf, (short)0, acc[c], false, false);
    }
  }

  // ---- add per-n / per-m (+b1) terms; C layout: row = hh*8+j, col h = c*16+r ----
#pragma unroll
  for (int c = 0; c < 8; ++c) {
    float py = sPY[wv * XP + c * 16 + r];
#pragma unroll
    for (int j = 0; j < 8; ++j) {
      int p = hh * 8 + j;
      acc[c][j] += sPX[p * XP + c * 16 + r] + py;
    }
  }

  // ---- fused LayerNorm + ReLU + W2 head + ReLU per pair row ----
  float ov[8];
#pragma unroll
  for (int j = 0; j < 8; ++j) {
    float s = 0.f, s2 = 0.f;
#pragma unroll
    for (int c = 0; c < 8; ++c) { float v = acc[c][j]; s += v; s2 += v * v; }
#pragma unroll
    for (int d = 1; d < 16; d <<= 1) {         // reduce over 16 h-columns in half
      s  += __shfl_xor(s,  d, 32);
      s2 += __shfl_xor(s2, d, 32);
    }
    const float mu  = s * (1.f / 128.f);
    const float var = s2 * (1.f / 128.f) - mu * mu;
    const float rs  = rsqrtf(var + 1e-5f);
    float o = 0.f;
#pragma unroll
    for (int c = 0; c < 8; ++c) {
      float v = (acc[c][j] - mu) * rs * gc[c] + bc[c];
      v = fmaxf(v, 0.f);
      o = fmaf(v, wc[c], o);
    }
#pragma unroll
    for (int d = 1; d < 16; d <<= 1) o += __shfl_xor(o, d, 32);
    ov[j] = fmaxf(o + b2v, 0.f);
  }

  // ---- store: pair p = hh*8 + j -> out[b, n0+p, m]; one writer lane per pair ----
#pragma unroll
  for (int j = 0; j < 8; ++j) {
    if (r == j) {
      int p = hh * 8 + j;
      out[((size_t)(bz * Nn + n0 + p)) * Mn + m] = ov[j];
    }
  }
}

extern "C" void kernel_launch(void* const* d_in, const int* in_sizes, int n_in,
                              void* d_out, int out_size, void* d_ws, size_t ws_size,
                              hipStream_t stream) {
  const float* x     = (const float*)d_in[0];
  const float* y     = (const float*)d_in[1];
  const float* W1    = (const float*)d_in[2];   // [384,128]
  const float* b1    = (const float*)d_in[3];
  const float* gamma = (const float*)d_in[4];
  const float* beta  = (const float*)d_in[5];
  const float* W2    = (const float*)d_in[6];
  const float* b2    = (const float*)d_in[7];
  float* out = (float*)d_out;

  char* ws = (char*)d_ws;
  float*    preX   = (float*)ws;                                  // 262144 f32
  float*    preY   = (float*)(ws + (size_t)262144 * 4);           // 262144 f32
  _Float16* w1d16  = (_Float16*)(ws + (size_t)2 * 262144 * 4);    // 16384 f16

  // Phase 1: per-n and per-m projections (+b1 folded into preY), W1d -> f16^T
  pre_gemm_kernel<<<Bn * Nn, 128, 0, stream>>>(x, W1,               b1, preX, 0);
  pre_gemm_kernel<<<Bn * Mn, 128, 0, stream>>>(y, W1 + Dn * Hn,     b1, preY, 1);
  w1d_cvt_kernel<<<64, 256, 0, stream>>>(W1, w1d16);

  // Phase 2: fused |x-y|@W1d + LN + head
  dim3 grid(Mn / 8, Nn / 16, Bn);   // (64, 32, 4)
  cost_kernel<<<grid, 256, 0, stream>>>(x, y, gamma, beta, W2, b2,
                                        preX, preY, w1d16, out);
}
